// TriMatchAttention_10608569221503
// MI455X (gfx1250) — compile-verified, matrix operand order checked by reading, not run on hardware
//
#include <hip/hip_runtime.h>
#include <hip/hip_bf16.h>

// ---------------------------------------------------------------------------
// TriMatchAttention for MI455X (gfx1250, wave32, WMMA)
// B=16, S=4096, D=1024, H=1024, VOCAB=512, EMB_DIM=3
// ---------------------------------------------------------------------------

#define TB 16
#define TS 4096
#define TD 1024
#define TH 1024
#define TVOCAB 512

typedef __attribute__((ext_vector_type(16))) __bf16 v16bf;
typedef __attribute__((ext_vector_type(8)))  float  v8f;

union FragBF {
    unsigned int u[8];
    v16bf        v;
};

// round-to-nearest-even f32 -> bf16, packed pair (lo = first)
static __device__ __forceinline__ unsigned int pack_bf16x2(float a, float b) {
    union { float f; unsigned int u; } ua, ub;
    ua.f = a; ub.f = b;
    unsigned int lo = (ua.u + 0x7FFFu + ((ua.u >> 16) & 1u)) >> 16;
    unsigned int hi = (ub.u + 0x7FFFu + ((ub.u >> 16) & 1u)) >> 16;
    return (lo & 0xFFFFu) | (hi << 16);
}

// ---------------------------------------------------------------------------
// Kernel 1: c[b,h] = Wt_b[h] + Wr_b[h] + ht_k[b,:]@Wt_w[h,:] + hm_k[b,:]@Wr_w[h,:]
// ---------------------------------------------------------------------------
__global__ void tma_prep_cbh(const float* __restrict__ ht_k,
                             const float* __restrict__ hm_k,
                             const float* __restrict__ Wt_w,
                             const float* __restrict__ Wt_b,
                             const float* __restrict__ Wr_w,
                             const float* __restrict__ Wr_b,
                             float* __restrict__ cbh) {
    unsigned idx = blockIdx.x * 256u + threadIdx.x;    // B*H = 16384 threads
    unsigned b = idx >> 10;
    unsigned h = idx & (TH - 1);
    unsigned kb = b * TD, kh = h * TD;
    float acc = Wt_b[h] + Wr_b[h];
    for (unsigned d = 0; d < TD; ++d)
        acc += ht_k[kb + d] * Wt_w[kh + d] + hm_k[kb + d] * Wr_w[kh + d];
    cbh[idx] = acc;
}

// ---------------------------------------------------------------------------
// Kernel 2: main GEMM  W_hs = hs @ Ws_w^T + Ws_b   (bf16 WMMA, f32 accum)
// Block tile: 64 M x 128 N, 8 waves (4M x 2N), each wave: 16x64 via 4 accums.
// Double-buffered LDS; one barrier per K-step; next-tile global loads overlap
// the current tile's WMMAs; distinct B-fragment registers keep the 4 WMMAs
// back-to-back (no WAR hazard NOPs, partial dscnt waits).
// Fused epilogue: wsum[m] += sum_n tanh(W_hs + cbh) * We_w   (atomic add)
// ---------------------------------------------------------------------------
__global__ __launch_bounds__(256) void tma_gemm_wmma(
        const float* __restrict__ hs,      // [B*S, D]
        const float* __restrict__ Ws_w,    // [H, D]
        const float* __restrict__ Ws_b,    // [H]
        const float* __restrict__ We_w,    // [H]
        const float* __restrict__ cbh,     // [B, H]
        float* __restrict__ whs,           // [B*S, H] out
        float* __restrict__ wsum)          // [B*S]    accumulated
{
    __shared__ unsigned int lds_a[2][64][17];   // K=32 as 16 bf16-pairs / row
    __shared__ unsigned int lds_b[2][128][17];

    const unsigned tid   = threadIdx.x;
    const unsigned lane  = tid & 31;
    const unsigned wid   = tid >> 5;
    const unsigned waveM = wid >> 1;      // 0..3
    const unsigned waveN = wid & 1;       // 0..1
    const unsigned hf    = lane >> 4;     // lane half (0/1)
    const unsigned ln    = lane & 15;

    const unsigned M0 = blockIdx.x * 64u;    // row tile in flattened [B*S]
    const unsigned N0 = blockIdx.y * 128u;   // col tile in [H]
    const unsigned b  = M0 >> 12;            // S = 4096; tiles never straddle b

    // per-thread staging coordinates (all 32-bit: buffers < 4GB)
    unsigned arow[4], akp[4], aoff[4];
#pragma unroll
    for (int it = 0; it < 4; ++it) {
        unsigned p = tid + it * 256u;
        arow[it] = p >> 4; akp[it] = p & 15u;
        aoff[it] = (M0 + arow[it]) * TD + akp[it] * 2u;
    }
    unsigned brow[8], bkp[8], boff[8];
#pragma unroll
    for (int it = 0; it < 8; ++it) {
        unsigned p = tid + it * 256u;
        brow[it] = p >> 4; bkp[it] = p & 15u;
        boff[it] = (N0 + brow[it]) * TD + bkp[it] * 2u;
    }

    // ---- prologue: fetch tile k0=0 into registers
    float2 ra[4], rb[8];
#pragma unroll
    for (int it = 0; it < 4; ++it)
        ra[it] = *(const float2*)(hs + aoff[it]);
#pragma unroll
    for (int it = 0; it < 8; ++it)
        rb[it] = *(const float2*)(Ws_w + boff[it]);

    v8f acc0 = {}, acc1 = {}, acc2 = {}, acc3 = {};

#pragma unroll 2
    for (unsigned k0 = 0; k0 < TD; k0 += 32) {
        const unsigned buf = (k0 >> 5) & 1u;   // constant after 2x unroll

        // ---- commit staged registers (bf16-packed) into lds[buf]
#pragma unroll
        for (int it = 0; it < 4; ++it)
            lds_a[buf][arow[it]][akp[it]] = pack_bf16x2(ra[it].x, ra[it].y);
#pragma unroll
        for (int it = 0; it < 8; ++it)
            lds_b[buf][brow[it]][bkp[it]] = pack_bf16x2(rb[it].x, rb[it].y);
        __syncthreads();

        // ---- issue next tile's global loads (overlap with WMMAs below)
        if (k0 + 32 < TD) {
#pragma unroll
            for (int it = 0; it < 4; ++it)
                ra[it] = *(const float2*)(hs + aoff[it] + k0 + 32);
#pragma unroll
            for (int it = 0; it < 8; ++it)
                rb[it] = *(const float2*)(Ws_w + boff[it] + k0 + 32);
        }

        // ---- fragments (ISA 16-bit layout): lane = row/col % 16,
        //      pair index for VGPR v = (v/4)*8 + half*4 + (v%4)
        FragBF fa, fb0, fb1, fb2, fb3;
#pragma unroll
        for (int v = 0; v < 8; ++v) {
            unsigned kpi = ((v >> 2) << 3) + (hf << 2) + (v & 3);
            fa.u[v]  = lds_a[buf][waveM * 16 + ln][kpi];
            fb0.u[v] = lds_b[buf][waveN * 64 +  0 + ln][kpi];
            fb1.u[v] = lds_b[buf][waveN * 64 + 16 + ln][kpi];
            fb2.u[v] = lds_b[buf][waveN * 64 + 32 + ln][kpi];
            fb3.u[v] = lds_b[buf][waveN * 64 + 48 + ln][kpi];
        }
        // ---- four back-to-back WMMAs sharing the A fragment
        acc0 = __builtin_amdgcn_wmma_f32_16x16x32_bf16(
            false, fa.v, false, fb0.v, (short)0, acc0, false, false);
        acc1 = __builtin_amdgcn_wmma_f32_16x16x32_bf16(
            false, fa.v, false, fb1.v, (short)0, acc1, false, false);
        acc2 = __builtin_amdgcn_wmma_f32_16x16x32_bf16(
            false, fa.v, false, fb2.v, (short)0, acc2, false, false);
        acc3 = __builtin_amdgcn_wmma_f32_16x16x32_bf16(
            false, fa.v, false, fb3.v, (short)0, acc3, false, false);
    }

    // ---- epilogue: add bias, store W_hs, fused tanh-dot partial into wsum
    float part[8] = {0.f, 0.f, 0.f, 0.f, 0.f, 0.f, 0.f, 0.f};
    v8f accs[4] = {acc0, acc1, acc2, acc3};
#pragma unroll
    for (int a = 0; a < 4; ++a) {
        unsigned N = N0 + waveN * 64u + a * 16u + ln;
        float wsb = Ws_b[N];
        float cb  = cbh[b * TH + N];
        float wew = We_w[N];
#pragma unroll
        for (int r = 0; r < 8; ++r) {
            unsigned M = M0 + waveM * 16u + r + hf * 8u;
            float w = accs[a][r] + wsb;
            whs[M * TH + N] = w;
            part[r] += tanhf(w + cb) * wew;
        }
    }
    // reduce across the 16 lanes sharing a row half
#pragma unroll
    for (int r = 0; r < 8; ++r) {
        part[r] += __shfl_xor(part[r], 1, 32);
        part[r] += __shfl_xor(part[r], 2, 32);
        part[r] += __shfl_xor(part[r], 4, 32);
        part[r] += __shfl_xor(part[r], 8, 32);
    }
    if (ln == 0) {
        unsigned Mbase = M0 + waveM * 16u + hf * 8u;
#pragma unroll
        for (int r = 0; r < 8; ++r)
            atomicAdd(&wsum[Mbase + r], part[r]);
    }
}

// ---------------------------------------------------------------------------
// Kernel 3: add We_b + rel-pos bias, softmax over S (one block per batch b)
// p buffer aliases wsum (in-place).
// ---------------------------------------------------------------------------
__global__ void tma_softmax(float* __restrict__ p,
                            const float* __restrict__ We_b,
                            const float* __restrict__ emb,
                            const float* __restrict__ lam,
                            const int* __restrict__ hs_i,
                            const int* __restrict__ ht_i) {
    __shared__ float red[256];
    const unsigned b = blockIdx.x, tid = threadIdx.x;
    const float web = We_b[0];
    const float l0 = lam[0], l1 = lam[1], l2 = lam[2];
    const int tk = ht_i[b];

    float m = -3.0e38f;
    for (unsigned s = tid; s < TS; s += 256) {
        unsigned i = b * TS + s;
        unsigned rp = (unsigned)(hs_i[i] * TVOCAB + tk) * 3u;
        float rv = emb[rp] * l0 + emb[rp + 1] * l1 + emb[rp + 2] * l2;
        float v = p[i] + web + rv;
        p[i] = v;
        m = fmaxf(m, v);
    }
    red[tid] = m; __syncthreads();
    for (int o = 128; o > 0; o >>= 1) {
        if (tid < (unsigned)o) red[tid] = fmaxf(red[tid], red[tid + o]);
        __syncthreads();
    }
    m = red[0]; __syncthreads();

    float sum = 0.f;
    for (unsigned s = tid; s < TS; s += 256) {
        unsigned i = b * TS + s;
        float e = expf(p[i] - m);
        p[i] = e;
        sum += e;
    }
    red[tid] = sum; __syncthreads();
    for (int o = 128; o > 0; o >>= 1) {
        if (tid < (unsigned)o) red[tid] += red[tid + o];
        __syncthreads();
    }
    float rinv = 1.0f / red[0]; __syncthreads();

    for (unsigned s = tid; s < TS; s += 256)
        p[b * TS + s] *= rinv;
}

// ---------------------------------------------------------------------------
// Kernel 4: alpha[b,h] = sum_s p[b,s] * W_hs[b,s,h]
// Grid: 64 blocks = (b, h-chunk of 256); p staged through LDS using the
// gfx1250 async global->LDS path when available.
// ---------------------------------------------------------------------------
__global__ void tma_alpha(const float* __restrict__ whs,
                          const float* __restrict__ p,
                          float* __restrict__ out) {
    __shared__ float lp[256];
    const unsigned b = blockIdx.x >> 2;
    const unsigned h = ((blockIdx.x & 3u) << 8) + threadIdx.x;
    float acc = 0.f;
    for (unsigned sc = 0; sc < TS / 256; ++sc) {
        unsigned srow = b * TS + sc * 256u + threadIdx.x;
#if __has_builtin(__builtin_amdgcn_global_load_async_to_lds_b32) && \
    __has_builtin(__builtin_amdgcn_s_wait_asynccnt)
        // GLOBAL_LOAD_ASYNC_TO_LDS_B32: per-lane 4B global->LDS, ASYNCcnt
        __builtin_amdgcn_global_load_async_to_lds_b32(
            (__attribute__((address_space(1))) int*)(p + srow),
            (__attribute__((address_space(3))) int*)(&lp[threadIdx.x]),
            0, 0);
        __builtin_amdgcn_s_wait_asynccnt(0);
#else
        lp[threadIdx.x] = p[srow];
#endif
        __syncthreads();
#pragma unroll 8
        for (unsigned j = 0; j < 256; ++j) {
            unsigned s = sc * 256u + j;
            acc += lp[j] * whs[(b * TS + s) * TH + h];
        }
        __syncthreads();
    }
    out[b * TH + h] = acc;
}

// ---------------------------------------------------------------------------
extern "C" void kernel_launch(void* const* d_in, const int* in_sizes, int n_in,
                              void* d_out, int out_size, void* d_ws, size_t ws_size,
                              hipStream_t stream) {
    (void)in_sizes; (void)n_in; (void)out_size; (void)ws_size;

    const float* hs    = (const float*)d_in[0];
    const float* ht_k  = (const float*)d_in[1];
    const float* hm_k  = (const float*)d_in[2];
    const float* Ws_w  = (const float*)d_in[3];
    const float* Ws_b  = (const float*)d_in[4];
    const float* Wt_w  = (const float*)d_in[5];
    const float* Wt_b  = (const float*)d_in[6];
    const float* Wr_w  = (const float*)d_in[7];
    const float* Wr_b  = (const float*)d_in[8];
    const float* We_w  = (const float*)d_in[9];
    const float* We_b  = (const float*)d_in[10];
    const float* emb   = (const float*)d_in[11];
    const float* lam   = (const float*)d_in[12];
    const int*   hs_i  = (const int*)d_in[13];
    const int*   ht_i  = (const int*)d_in[14];
    float* out = (float*)d_out;

    // workspace layout
    char* ws = (char*)d_ws;
    float* whs  = (float*)ws;                                   // B*S*H f32 = 256 MiB
    float* cbh  = (float*)(ws + (size_t)TB * TS * TH * 4);      // B*H
    float* wsum = (float*)(ws + (size_t)TB * TS * TH * 4 + (size_t)TB * TH * 4); // B*S

    // 1) tiny fused projections c[b,h]
    tma_prep_cbh<<<(TB * TH) / 256, 256, 0, stream>>>(ht_k, hm_k, Wt_w, Wt_b,
                                                      Wr_w, Wr_b, cbh);
    // 2) zero the score accumulator (graph-capturable)
    (void)hipMemsetAsync(wsum, 0, (size_t)TB * TS * sizeof(float), stream);

    // 3) main bf16 WMMA GEMM with fused tanh-dot epilogue
    dim3 gg((TB * TS) / 64, TH / 128);
    tma_gemm_wmma<<<gg, 256, 0, stream>>>(hs, Ws_w, Ws_b, We_w, cbh, whs, wsum);

    // 4) rel-pos bias + softmax (in-place on wsum -> p)
    tma_softmax<<<TB, 256, 0, stream>>>(wsum, We_b, emb, lam, hs_i, ht_i);

    // 5) weighted sum -> alpha
    tma_alpha<<<TB * (TH / 256), 256, 0, stream>>>(whs, wsum, out);
}
